// EHM_83769042141613
// MI455X (gfx1250) — matrix operand order repaired and checked by reading, not compiled, not run on hardware
//
#include <hip/hip_runtime.h>
#include <hip/hip_bf16.h>
#include <math.h>

// ---------------------------------------------------------------------------
// EHM (SMPL-X + FLAME + MANO LBS) pipeline for gfx1250 (MI455X, wave32, WMMA)
// Big GEMMs (shapedirs / posedirs) use v_wmma_f32_16x16x32_f16 (f16 in, f32 acc)
// with 128x32x64 block tiles: 8 waves x (2 n-tiles x 2 k-steps) = 4 WMMA/stage.
// ---------------------------------------------------------------------------

typedef __attribute__((ext_vector_type(16))) _Float16 v16h;
typedef __attribute__((ext_vector_type(8)))  _Float16 v8h;
typedef __attribute__((ext_vector_type(4)))  _Float16 v4h;
typedef __attribute__((ext_vector_type(2)))  _Float16 v2h;
typedef __attribute__((ext_vector_type(8)))  float    v8f;

#define BB   128      // batch
#define VS_  10475
#define VF_  5023
#define VM_  778
#define NSE  350      // NS + NE
#define KP_NSE 384    // NSE padded to 64
#define KP_MB  64     // 10  padded
#define KP_PF  64     // 36  padded
#define KP_PM  192    // 135 padded
#define KP_PS  512    // 486 padded
#define NJS  55
#define NJF  5
#define NJM  16

__constant__ int c_par_smplx[NJS] = {
  -1, 0, 0, 0, 1, 2, 3, 4, 5, 6, 7, 8, 9, 9, 9, 12, 13, 14, 16, 17, 18, 19,
  15, 15, 15, 20, 25, 26, 20, 28, 29, 20, 31, 32, 20, 34, 35, 20, 37, 38, 21,
  40, 41, 21, 43, 44, 21, 46, 47, 21, 49, 50, 21, 52, 53};
__constant__ int c_par_flame[NJF] = {-1, 0, 1, 1, 1};
__constant__ int c_par_mano[NJM]  = {-1, 0, 1, 2, 0, 4, 5, 0, 7, 8, 0, 10, 11, 0, 13, 14};

// ---------------------------------------------------------------------------
// WMMA GEMM: out[b,n] = (src?src[b,n]:0) + (base?base[n]:0) + sum_k A[b,k]*B(k,n)
//   A : 128 x Kp row-major f32, zero-padded for k in [K,Kp), Kp % 64 == 0
//   Bm: bNK==1 -> (N x K) row-major (shapedirs); bNK==0 -> (K x N) (posedirs)
// Block: 256 thr = 8 waves; wave w owns M rows [16w,16w+16); block owns 32 N cols.
// ---------------------------------------------------------------------------
__global__ __launch_bounds__(256)
void ehm_wmma_gemm(const float* __restrict__ A, const float* __restrict__ Bm,
                   const float* __restrict__ base, const float* __restrict__ src,
                   float* __restrict__ out, int N, int K, int Kp, int bNK)
{
  const int tid  = threadIdx.x;
  const int wave = tid >> 5;
  const int lane = tid & 31;
  const int n0   = blockIdx.x * 32;

  __shared__ _Float16 Asl[128 * 64];  // 16 KB, row-major [r][k]
  __shared__ _Float16 Bsl[32 * 64];   // 4 KB,  n-major  [n][k]

  v8f acc0 = {};
  v8f acc1 = {};

  const int m   = lane & 15;
  const int kbA = (lane < 16) ? 0 : 16;   // A: lanes>=16 hold K 16..31
  const int kbB = (lane < 16) ? 0 : 16;   // B: lanes>=16 hold K 16..31

  for (int k0 = 0; k0 < Kp; k0 += 64) {
    // ---- stage A tile: 128 rows x 64 k, branchless float4 -> v4h
    for (int i = tid; i < 128 * 16; i += 256) {
      int r = i >> 4, c4 = (i & 15) << 2;
      const float4 f = *(const float4*)(A + (size_t)r * Kp + k0 + c4);
      v4h h = {(_Float16)f.x, (_Float16)f.y, (_Float16)f.z, (_Float16)f.w};
      *(v4h*)&Asl[r * 64 + c4] = h;
    }
    // ---- stage B tile: 64 k x 32 n into n-major LDS
    const bool fullTile = (n0 + 32 <= N) && (k0 + 64 <= K);
    if (fullTile) {
      if (bNK) {
        // (N x K), K even: 2 contiguous k per thread as float2 -> one b32 store
        for (int i = tid; i < 1024; i += 256) {
          int nn = i >> 5, k2 = (i & 31) << 1;
          const float2 f = *(const float2*)(Bm + (size_t)(n0 + nn) * K + k0 + k2);
          v2h h = {(_Float16)f.x, (_Float16)f.y};
          *(v2h*)&Bsl[nn * 64 + k2] = h;
        }
      } else {
        // (K x N): 2 contiguous n per thread (scalar loads, transposed stores)
        for (int i = tid; i < 1024; i += 256) {
          int kk = i >> 4, n2 = (i & 15) << 1;
          const float* p = Bm + (size_t)(k0 + kk) * N + n0 + n2;
          Bsl[(n2 + 0) * 64 + kk] = (_Float16)p[0];
          Bsl[(n2 + 1) * 64 + kk] = (_Float16)p[1];
        }
      }
    } else if (bNK) {
      for (int i = tid; i < 2048; i += 256) {        // contiguous k per thread run
        int nn = i >> 6, kk = i & 63;
        int k = k0 + kk, gn = n0 + nn;
        float v = (k < K && gn < N) ? Bm[(size_t)gn * K + k] : 0.0f;
        Bsl[nn * 64 + kk] = (_Float16)v;
      }
    } else {
      for (int i = tid; i < 2048; i += 256) {        // contiguous n per thread run
        int kk = i >> 5, nn = i & 31;
        int k = k0 + kk, gn = n0 + nn;
        float v = (k < K && gn < N) ? Bm[(size_t)k * N + gn] : 0.0f;
        Bsl[nn * 64 + kk] = (_Float16)v;
      }
    }
    __syncthreads();

#pragma unroll
    for (int ks = 0; ks < 64; ks += 32) {
      // A fragment: halves h -> K = kbA + h (16 contiguous halves)
      const _Float16* ar = &Asl[(wave * 16 + m) * 64 + ks];
      v8h alo = *(const v8h*)(ar + kbA);
      v8h ahi = *(const v8h*)(ar + kbA + 8);
      v16h af = __builtin_shufflevector(alo, ahi, 0, 1, 2, 3, 4, 5, 6, 7,
                                        8, 9, 10, 11, 12, 13, 14, 15);
      // B fragments: halves h -> K = kbB + h (16 contiguous halves)
      const _Float16* br0 = &Bsl[m * 64 + ks + kbB];
      v8h b0l = *(const v8h*)(br0);
      v8h b0h = *(const v8h*)(br0 + 8);
      v16h bf0 = __builtin_shufflevector(b0l, b0h, 0, 1, 2, 3, 4, 5, 6, 7,
                                         8, 9, 10, 11, 12, 13, 14, 15);
      const _Float16* br1 = &Bsl[(16 + m) * 64 + ks + kbB];
      v8h b1l = *(const v8h*)(br1);
      v8h b1h = *(const v8h*)(br1 + 8);
      v16h bf1 = __builtin_shufflevector(b1l, b1h, 0, 1, 2, 3, 4, 5, 6, 7,
                                         8, 9, 10, 11, 12, 13, 14, 15);

      acc0 = __builtin_amdgcn_wmma_f32_16x16x32_f16(false, af, false, bf0,
                                                    (short)0, acc0, false, false);
      acc1 = __builtin_amdgcn_wmma_f32_16x16x32_f16(false, af, false, bf1,
                                                    (short)0, acc1, false, false);
    }
    __syncthreads();
  }

  // C/D layout: lane n = lane%16; VGPR r -> M = r + (lane<16 ? 0 : 8)
  const int mofs = (lane < 16) ? 0 : 8;
#pragma unroll
  for (int t = 0; t < 2; ++t) {
    int gn = n0 + t * 16 + m;
    if (gn < N) {
      const float basev = base ? base[gn] : 0.0f;
      const v8f accv = t ? acc1 : acc0;
#pragma unroll
      for (int r = 0; r < 8; ++r) {
        int gm = wave * 16 + r + mofs;
        size_t o = (size_t)gm * N + gn;
        float v = accv[r] + basev;
        if (src) v += src[o];
        out[o] = v;
      }
    }
  }
}

// ---------------------------------------------------------------------------
// small-op kernels
// ---------------------------------------------------------------------------
__global__ void ehm_fill0(float* p, int n)
{
  int i = blockIdx.x * blockDim.x + threadIdx.x;
  if (i < n) p[i] = 0.0f;
}

__global__ void ehm_prep(const float* shp, const float* fexp, const float* eye,
                         const float* jaw, const float* bexp, const float* gpose,
                         const float* bpose, const float* lhp, const float* rhp,
                         const float* manob,
                         float* betas_f, float* sc, float* pose_f, float* pose_l,
                         float* pose_r, float* full_pose, float* mb)
{
  int b = blockIdx.x * blockDim.x + threadIdx.x;
  if (b >= BB) return;
  for (int l = 0; l < 300; ++l) { betas_f[b*KP_NSE+l] = shp[b*300+l]; sc[b*KP_NSE+l] = shp[b*300+l]; }
  for (int l = 0; l < 50; ++l)  { betas_f[b*KP_NSE+300+l] = fexp[b*50+l]; sc[b*KP_NSE+300+l] = bexp[b*50+l]; }
  for (int i = 0; i < 6; ++i) pose_f[b*15+i] = 0.0f;
  for (int i = 0; i < 3; ++i) pose_f[b*15+6+i] = jaw[b*3+i];
  for (int i = 0; i < 6; ++i) pose_f[b*15+9+i] = eye[b*6+i];
  for (int i = 0; i < 3; ++i)  { pose_l[b*48+i] = 0.0f; pose_r[b*48+i] = 0.0f; }
  for (int i = 0; i < 45; ++i) { pose_l[b*48+3+i] = lhp[b*45+i]; pose_r[b*48+3+i] = rhp[b*45+i]; }
  for (int i = 0; i < 3; ++i)  full_pose[b*165+i] = gpose[b*3+i];
  for (int i = 0; i < 63; ++i) full_pose[b*165+3+i] = bpose[b*63+i];
  for (int i = 66; i < 165; ++i) full_pose[b*165+i] = 0.0f;
  for (int i = 0; i < 10; ++i) mb[b*KP_MB+i] = manob[i];
}

// rot: (B,Nj,9); pf: (B, pfstride) with (j-1)*9+t columns (pre-zeroed padding)
__global__ void ehm_rodrigues(const float* __restrict__ pose, float* __restrict__ rot,
                              float* __restrict__ pf, int Nj, int pfstride)
{
  int i = blockIdx.x * blockDim.x + threadIdx.x;
  if (i >= BB * Nj) return;
  int j = i % Nj;
  float x = pose[i*3+0], y = pose[i*3+1], z = pose[i*3+2];
  float ang = sqrtf(x*x + y*y + z*z + 1e-8f);
  float inv = 1.0f / ang;
  float ux = x*inv, uy = y*inv, uz = z*inv;
  float s = sinf(ang), c = cosf(ang);
  float Km[9] = {0.f,-uz, uy,  uz, 0.f,-ux,  -uy, ux, 0.f};
  float K2[9];
#pragma unroll
  for (int r = 0; r < 3; ++r)
#pragma unroll
    for (int q = 0; q < 3; ++q)
      K2[r*3+q] = Km[r*3+0]*Km[0*3+q] + Km[r*3+1]*Km[1*3+q] + Km[r*3+2]*Km[2*3+q];
#pragma unroll
  for (int t = 0; t < 9; ++t) {
    float id = (t == 0 || t == 4 || t == 8) ? 1.0f : 0.0f;
    float R = id + s * Km[t] + (1.0f - c) * K2[t];
    rot[(size_t)i*9 + t] = R;
    if (j > 0) {
      int bb = i / Nj;
      pf[(size_t)bb * pfstride + (j-1)*9 + t] = R - id;
    }
  }
}

// J[b,j,c] = sum_v reg[j,v] * vsh[b,v,c]  (+ joff)
__global__ __launch_bounds__(128)
void ehm_jreg(const float* __restrict__ reg, const float* __restrict__ vsh,
              const float* __restrict__ joff, float* __restrict__ Jout, int V, int J)
{
  int j = blockIdx.x, b = blockIdx.y, tid = threadIdx.x;
  float s0 = 0.f, s1 = 0.f, s2 = 0.f;
  for (int v = tid; v < V; v += 128) {
    float w = reg[(size_t)j * V + v];
    const float* p = vsh + ((size_t)b * V + v) * 3;
    s0 += w * p[0]; s1 += w * p[1]; s2 += w * p[2];
  }
  __shared__ float red[3][128];
  red[0][tid] = s0; red[1][tid] = s1; red[2][tid] = s2;
  __syncthreads();
  for (int off = 64; off > 0; off >>= 1) {
    if (tid < off) {
      red[0][tid] += red[0][tid+off];
      red[1][tid] += red[1][tid+off];
      red[2][tid] += red[2][tid+off];
    }
    __syncthreads();
  }
  if (tid == 0) {
    size_t o = ((size_t)b * J + j) * 3;
    float o0 = joff ? joff[o+0] : 0.f, o1 = joff ? joff[o+1] : 0.f, o2 = joff ? joff[o+2] : 0.f;
    Jout[o+0] = red[0][0] + o0; Jout[o+1] = red[1][0] + o1; Jout[o+2] = red[2][0] + o2;
  }
}

// forward kinematics: one thread per batch element
__global__ void ehm_fk(const float* __restrict__ rot, const float* __restrict__ J,
                       float* __restrict__ A, float* __restrict__ posedJ,
                       int Nj, int which)
{
  int b = blockIdx.x * blockDim.x + threadIdx.x;
  if (b >= BB) return;
  const int* par = (which == 0) ? c_par_smplx : (which == 1) ? c_par_flame : c_par_mano;
  for (int j = 0; j < Nj; ++j) {
    int p = par[j];
    float rel[3];
#pragma unroll
    for (int c = 0; c < 3; ++c) {
      float jj = J[((size_t)b*Nj + j)*3 + c];
      rel[c] = (p >= 0) ? (jj - J[((size_t)b*Nj + p)*3 + c]) : jj;
    }
    float T[16];
#pragma unroll
    for (int r = 0; r < 3; ++r) {
#pragma unroll
      for (int q = 0; q < 3; ++q) T[r*4+q] = rot[((size_t)b*Nj + j)*9 + r*3 + q];
      T[r*4+3] = rel[r];
    }
    T[12] = 0.f; T[13] = 0.f; T[14] = 0.f; T[15] = 1.f;
    float Ao[16];
    if (p < 0) {
#pragma unroll
      for (int t = 0; t < 16; ++t) Ao[t] = T[t];
    } else {
      const float* Ap = A + ((size_t)b*Nj + p)*16;
#pragma unroll
      for (int r = 0; r < 4; ++r)
#pragma unroll
        for (int q = 0; q < 4; ++q)
          Ao[r*4+q] = Ap[r*4+0]*T[0*4+q] + Ap[r*4+1]*T[1*4+q]
                    + Ap[r*4+2]*T[2*4+q] + Ap[r*4+3]*T[3*4+q];
    }
    float* Aj = A + ((size_t)b*Nj + j)*16;
#pragma unroll
    for (int t = 0; t < 16; ++t) Aj[t] = Ao[t];
#pragma unroll
    for (int c = 0; c < 3; ++c) posedJ[((size_t)b*Nj + j)*3 + c] = Ao[c*4+3];
  }
  // A_rel: A[:, :3, 3] -= A[:, :3, :3] @ J[j]
  for (int j = 0; j < Nj; ++j) {
    float* Aj = A + ((size_t)b*Nj + j)*16;
    float jx = J[((size_t)b*Nj + j)*3+0], jy = J[((size_t)b*Nj + j)*3+1], jz = J[((size_t)b*Nj + j)*3+2];
#pragma unroll
    for (int r = 0; r < 3; ++r)
      Aj[r*4+3] -= Aj[r*4+0]*jx + Aj[r*4+1]*jy + Aj[r*4+2]*jz;
  }
}

// skinning (+ optional eyelids, scale, offset)
__global__ __launch_bounds__(256)
void ehm_skin(const float* __restrict__ vposed, const float* __restrict__ A,
              const float* __restrict__ W, float* __restrict__ out, int V, int J,
              const float* __restrict__ scale, const float* __restrict__ offs,
              const float* __restrict__ reye, const float* __restrict__ leye,
              const float* __restrict__ epar)
{
  int b = blockIdx.y;
  int v = blockIdx.x * blockDim.x + threadIdx.x;
  __shared__ float As[NJS * 16];
  for (int i = threadIdx.x; i < J * 16; i += blockDim.x)
    As[i] = A[(size_t)b * J * 16 + i];
  __syncthreads();
  if (v >= V) return;
  float T[12] = {0,0,0,0,0,0,0,0,0,0,0,0};
  const float* wr = W + (size_t)v * J;
  for (int j = 0; j < J; ++j) {
    float w = wr[j];
    const float* a = As + j * 16;
#pragma unroll
    for (int t = 0; t < 12; ++t) T[t] += w * a[t];
  }
  const float* vp = vposed + ((size_t)b * V + v) * 3;
  float x = vp[0], y = vp[1], z = vp[2];
  float ox = T[0]*x + T[1]*y + T[2]*z  + T[3];
  float oy = T[4]*x + T[5]*y + T[6]*z  + T[7];
  float oz = T[8]*x + T[9]*y + T[10]*z + T[11];
  if (reye) {
    float e0 = epar[b*2+0], e1 = epar[b*2+1];
    ox += reye[(size_t)v*3+0]*e1 + leye[(size_t)v*3+0]*e0;
    oy += reye[(size_t)v*3+1]*e1 + leye[(size_t)v*3+1]*e0;
    oz += reye[(size_t)v*3+2]*e1 + leye[(size_t)v*3+2]*e0;
  }
  if (scale) {
    float s = scale[b];
    ox = ox*s + offs[b*3+0]; oy = oy*s + offs[b*3+1]; oz = oz*s + offs[b*3+2];
  }
  size_t o = ((size_t)b * V + v) * 3;
  out[o+0] = ox; out[o+1] = oy; out[o+2] = oz;
}

// head replacement: new_t[b, s2f[hi]] = head_v[b,hi] - mean(headJ[3:5]) + mean(tbj[23:25])
__global__ void ehm_head_scatter(const float* __restrict__ head_v, const float* __restrict__ pJf,
                                 const float* __restrict__ tbj, const int* __restrict__ head_index,
                                 const int* __restrict__ s2f, float* __restrict__ new_t, int nIdx)
{
  int b = blockIdx.y;
  int i = blockIdx.x * blockDim.x + threadIdx.x;
  if (i >= nIdx) return;
  int hi = head_index[i];
  int v  = s2f[hi];
#pragma unroll
  for (int c = 0; c < 3; ++c) {
    float hj = 0.5f * (pJf[((size_t)b*NJF + 3)*3 + c] + pJf[((size_t)b*NJF + 4)*3 + c]);
    float tj = 0.5f * (tbj[((size_t)b*NJS + 23)*3 + c] + tbj[((size_t)b*NJS + 24)*3 + c]);
    new_t[((size_t)b*VS_ + v)*3 + c] = head_v[((size_t)b*VF_ + hi)*3 + c] - hj + tj;
  }
}

// hand replacement: new_t[b, map[i]] = sgn*(hv[b,i] - posedJ0) + tbj[jidx]
__global__ void ehm_hand_scatter(const float* __restrict__ hv, const float* __restrict__ pJm,
                                 const float* __restrict__ tbj, const int* __restrict__ map,
                                 float* __restrict__ new_t, int jidx, float sx)
{
  int b = blockIdx.y;
  int i = blockIdx.x * blockDim.x + threadIdx.x;
  if (i >= VM_) return;
  int v = map[i];
  float sg[3] = {sx, 1.0f, 1.0f};
#pragma unroll
  for (int c = 0; c < 3; ++c) {
    float val = hv[((size_t)b*VM_ + i)*3 + c] - pJm[((size_t)b*NJM + 0)*3 + c];
    new_t[((size_t)b*VS_ + v)*3 + c] = sg[c]*val + tbj[((size_t)b*NJS + jidx)*3 + c];
  }
}

// ---------------------------------------------------------------------------
extern "C" void kernel_launch(void* const* d_in, const int* in_sizes, int n_in,
                              void* d_out, int out_size, void* d_ws, size_t ws_size,
                              hipStream_t stream)
{
  const float* shape_params = (const float*)d_in[0];
  const float* flame_exp    = (const float*)d_in[1];
  const float* eye_pose     = (const float*)d_in[2];
  const float* jaw_params   = (const float*)d_in[3];
  const float* eyelid_par   = (const float*)d_in[4];
  const float* head_scale   = (const float*)d_in[5];
  const float* head_off     = (const float*)d_in[6];
  const float* lh_pose      = (const float*)d_in[7];
  const float* rh_pose      = (const float*)d_in[8];
  const float* lh_scale     = (const float*)d_in[9];
  const float* lh_off       = (const float*)d_in[10];
  const float* rh_scale     = (const float*)d_in[11];
  const float* rh_off       = (const float*)d_in[12];
  const float* body_exp     = (const float*)d_in[13];
  const float* global_pose  = (const float*)d_in[14];
  const float* body_pose    = (const float*)d_in[15];
  const float* joints_off   = (const float*)d_in[16];
  const float* mano_betas   = (const float*)d_in[17];
  const float* f_vtmpl      = (const float*)d_in[18];
  const float* f_shape      = (const float*)d_in[19];
  const float* f_pose       = (const float*)d_in[20];
  const float* f_jreg       = (const float*)d_in[21];
  const float* f_W          = (const float*)d_in[22];
  const float* r_eyelid     = (const float*)d_in[23];
  const float* l_eyelid     = (const float*)d_in[24];
  const float* m_vtmpl      = (const float*)d_in[25];
  const float* m_shape      = (const float*)d_in[26];
  const float* m_pose       = (const float*)d_in[27];
  const float* m_jreg       = (const float*)d_in[28];
  const float* m_W          = (const float*)d_in[29];
  const float* s_vtmpl      = (const float*)d_in[30];
  const float* s_shape      = (const float*)d_in[31];
  const float* s_pose       = (const float*)d_in[32];
  const float* s_jreg       = (const float*)d_in[33];
  const float* s_W          = (const float*)d_in[34];
  const int*   s2f          = (const int*)d_in[35];
  const int*   head_index   = (const int*)d_in[36];
  const int*   s2ml         = (const int*)d_in[37];
  const int*   s2mr         = (const int*)d_in[38];
  const int    nHead        = in_sizes[36];

  float* ws = (float*)d_ws;
  size_t off = 0;
  auto alloc = [&](size_t n) { float* p = ws + off; off += n; return p; };

  // --- padded A-operand group first (one contiguous zero-fill covers all) ---
  float* betas_f  = alloc((size_t)BB * KP_NSE);
  float* sc       = alloc((size_t)BB * KP_NSE);
  float* mb       = alloc((size_t)BB * KP_MB);
  float* pf_f     = alloc((size_t)BB * KP_PF);
  float* pf_l     = alloc((size_t)BB * KP_PM);
  float* pf_r     = alloc((size_t)BB * KP_PM);
  float* pf_s     = alloc((size_t)BB * KP_PS);
  const size_t aGroup = off;   // elements to zero-fill

  float* pose_f   = alloc((size_t)BB * 15);
  float* pose_l   = alloc((size_t)BB * 48);
  float* pose_r   = alloc((size_t)BB * 48);
  float* full_p   = alloc((size_t)BB * 165);
  float* rot_f    = alloc((size_t)BB * NJF * 9);
  float* rot_l    = alloc((size_t)BB * NJM * 9);
  float* rot_r    = alloc((size_t)BB * NJM * 9);
  float* rot_s    = alloc((size_t)BB * NJS * 9);
  float* vsh_f    = alloc((size_t)BB * VF_ * 3);   // -> v_posed_f -> head_v (in-place)
  float* J_f      = alloc((size_t)BB * NJF * 3);
  float* pJ_f     = alloc((size_t)BB * NJF * 3);
  float* A_f      = alloc((size_t)BB * NJF * 16);
  float* vsh_m    = alloc((size_t)BB * VM_ * 3);   // shared rest shape for both hands
  float* J_m      = alloc((size_t)BB * NJM * 3);
  float* vpos_l   = alloc((size_t)BB * VM_ * 3);   // -> lv (in-place skin)
  float* vpos_r   = alloc((size_t)BB * VM_ * 3);   // -> rv
  float* pJ_l     = alloc((size_t)BB * NJM * 3);
  float* A_l      = alloc((size_t)BB * NJM * 16);
  float* pJ_r     = alloc((size_t)BB * NJM * 3);
  float* A_r      = alloc((size_t)BB * NJM * 16);
  float* new_t    = alloc((size_t)BB * VS_ * 3);   // -> v_posed (in-place)
  float* tbj      = alloc((size_t)BB * NJS * 3);
  float* pJ_s     = alloc((size_t)BB * NJS * 3);
  float* A_s      = alloc((size_t)BB * NJS * 16);
  (void)off; (void)ws_size; (void)n_in; (void)out_size;

  const int NF = VF_ * 3, NM = VM_ * 3, NSX = VS_ * 3;

  // 0) zero the padded A-operand region (pads must be 0 for unguarded WMMA loads)
  ehm_fill0<<<(int)((aGroup + 255) / 256), 256, 0, stream>>>(ws, (int)aGroup);

  // 1) concat / broadcast small params
  ehm_prep<<<1, BB, 0, stream>>>(shape_params, flame_exp, eye_pose, jaw_params,
                                 body_exp, global_pose, body_pose, lh_pose, rh_pose,
                                 mano_betas, betas_f, sc, pose_f, pose_l, pose_r,
                                 full_p, mb);

  // 2) rodrigues for all pose sets (pf written with padded stride)
  ehm_rodrigues<<<(BB*NJF + 255)/256, 256, 0, stream>>>(pose_f, rot_f, pf_f, NJF, KP_PF);
  ehm_rodrigues<<<(BB*NJM + 255)/256, 256, 0, stream>>>(pose_l, rot_l, pf_l, NJM, KP_PM);
  ehm_rodrigues<<<(BB*NJM + 255)/256, 256, 0, stream>>>(pose_r, rot_r, pf_r, NJM, KP_PM);
  ehm_rodrigues<<<(BB*NJS + 255)/256, 256, 0, stream>>>(full_p, rot_s, pf_s, NJS, KP_PS);

  // 3) shapedirs GEMMs (WMMA): out = template + betas @ shapedirs^T
  ehm_wmma_gemm<<<(NF  + 31)/32, 256, 0, stream>>>(betas_f, f_shape, f_vtmpl, nullptr, vsh_f, NF,  NSE, KP_NSE, 1);
  ehm_wmma_gemm<<<(NM  + 31)/32, 256, 0, stream>>>(mb,      m_shape, m_vtmpl, nullptr, vsh_m, NM,  10,  KP_MB,  1);
  ehm_wmma_gemm<<<(NSX + 31)/32, 256, 0, stream>>>(sc,      s_shape, s_vtmpl, nullptr, new_t, NSX, NSE, KP_NSE, 1);

  // 4) joint regressors (tbj includes joints_offset)
  { dim3 g(NJF, BB); ehm_jreg<<<g, 128, 0, stream>>>(f_jreg, vsh_f, nullptr,    J_f, VF_, NJF); }
  { dim3 g(NJM, BB); ehm_jreg<<<g, 128, 0, stream>>>(m_jreg, vsh_m, nullptr,    J_m, VM_, NJM); }
  { dim3 g(NJS, BB); ehm_jreg<<<g, 128, 0, stream>>>(s_jreg, new_t, joints_off, tbj, VS_, NJS); }

  // 5) posedirs GEMMs (WMMA accumulate): flame in-place; hands from shared rest shape
  ehm_wmma_gemm<<<(NF + 31)/32, 256, 0, stream>>>(pf_f, f_pose, nullptr, vsh_f, vsh_f,  NF, 36,  KP_PF, 0);
  ehm_wmma_gemm<<<(NM + 31)/32, 256, 0, stream>>>(pf_l, m_pose, nullptr, vsh_m, vpos_l, NM, 135, KP_PM, 0);
  ehm_wmma_gemm<<<(NM + 31)/32, 256, 0, stream>>>(pf_r, m_pose, nullptr, vsh_m, vpos_r, NM, 135, KP_PM, 0);

  // 6) forward kinematics (head + hands)
  ehm_fk<<<1, BB, 0, stream>>>(rot_f, J_f, A_f, pJ_f, NJF, 1);
  ehm_fk<<<1, BB, 0, stream>>>(rot_l, J_m, A_l, pJ_l, NJM, 2);
  ehm_fk<<<1, BB, 0, stream>>>(rot_r, J_m, A_r, pJ_r, NJM, 2);

  // 7) skin head (fused eyelids + scale + offset) and hands (fused scale + offset)
  { dim3 g((VF_ + 255)/256, BB);
    ehm_skin<<<g, 256, 0, stream>>>(vsh_f, A_f, f_W, vsh_f, VF_, NJF,
                                    head_scale, head_off, r_eyelid, l_eyelid, eyelid_par); }
  { dim3 g((VM_ + 255)/256, BB);
    ehm_skin<<<g, 256, 0, stream>>>(vpos_l, A_l, m_W, vpos_l, VM_, NJM,
                                    lh_scale, lh_off, nullptr, nullptr, nullptr);
    ehm_skin<<<g, 256, 0, stream>>>(vpos_r, A_r, m_W, vpos_r, VM_, NJM,
                                    rh_scale, rh_off, nullptr, nullptr, nullptr); }

  // 8) scatter head / hand vertices into SMPL-X template (after tbj, before posedirs)
  { dim3 g((nHead + 255)/256, BB);
    ehm_head_scatter<<<g, 256, 0, stream>>>(vsh_f, pJ_f, tbj, head_index, s2f, new_t, nHead); }
  { dim3 g((VM_ + 255)/256, BB);
    ehm_hand_scatter<<<g, 256, 0, stream>>>(vpos_l, pJ_l, tbj, s2ml, new_t, 20, -1.0f);
    ehm_hand_scatter<<<g, 256, 0, stream>>>(vpos_r, pJ_r, tbj, s2mr, new_t, 21,  1.0f); }

  // 9) SMPL-X posedirs (WMMA, in-place accumulate into new_t -> v_posed)
  ehm_wmma_gemm<<<(NSX + 31)/32, 256, 0, stream>>>(pf_s, s_pose, nullptr, new_t, new_t, NSX, 486, KP_PS, 0);

  // 10) SMPL-X FK
  ehm_fk<<<1, BB, 0, stream>>>(rot_s, tbj, A_s, pJ_s, NJS, 0);

  // 11) final skinning -> d_out (B, VS, 3)
  { dim3 g((VS_ + 255)/256, BB);
    ehm_skin<<<g, 256, 0, stream>>>(new_t, A_s, s_W, (float*)d_out, VS_, NJS,
                                    nullptr, nullptr, nullptr, nullptr, nullptr); }
}
